// TinyViTBlock_47588237640049
// MI455X (gfx1250) — matrix operand order repaired and checked by reading, not compiled, not run on hardware
//
#include <hip/hip_runtime.h>
#include <hip/hip_bf16.h>

typedef __bf16 bf16;
typedef __attribute__((ext_vector_type(16))) __bf16 v16bf;
typedef __attribute__((ext_vector_type(8)))  __bf16 v8bf;
typedef __attribute__((ext_vector_type(8)))  float  v8f;

#define DIM 384
#define HEADS 12
#define HD 32
#define WS7 7
#define NTOK 49
#define HIDDEN 1536
#define SCALE_ATT 0.17677669529663687f

// ---------------- fragment helpers (CDNA5 WMMA layouts, ISA 7.12.2) -------

static __device__ __forceinline__ v8f vzero() {
  v8f z;
#pragma unroll
  for (int i = 0; i < 8; i++) z[i] = 0.f;
  return z;
}

// assemble v16bf from two 8-element (16B) chunks
static __device__ __forceinline__ v16bf frag2(const bf16* p0, const bf16* p1) {
  v8bf lo = *(const v8bf*)p0;
  v8bf hi = *(const v8bf*)p1;
  v16bf r;
#pragma unroll
  for (int i = 0; i < 8; i++) { r[i] = lo[i]; r[i + 8] = hi[i]; }
  return r;
}

// A fragment 16x32 bf16: lane m = lane&15, chunks at k0=(lane>>4)*8 and k0+16
static __device__ __forceinline__ v16bf loadA(const bf16* rowptr_at_koff) {
  return frag2(rowptr_at_koff, rowptr_at_koff + 16);
}
// B fragment 32x16 bf16: lane col = lane&15, chunks at k0=(lane>>4)*8 and k0+16
static __device__ __forceinline__ v16bf loadB(const bf16* p) {
  return frag2(p, p + 16);
}

static __device__ __forceinline__ v8f wmma_bf16(v16bf a, v16bf b, v8f c) {
  return __builtin_amdgcn_wmma_f32_16x16x32_bf16(false, a, false, b,
                                                 (short)0, c, false, false);
}

// ---------------- prep kernels --------------------------------------------

__global__ void cvt_bf16_kernel(const float* __restrict__ src,
                                bf16* __restrict__ dst, int n) {
  int i = blockIdx.x * 256 + threadIdx.x;
  if (i < n) dst[i] = (bf16)src[i];
}

// bias laid out in WMMA C-fragment order: [h][mt][nt][jv][lane]
__global__ void bias_frag_kernel(const float* __restrict__ bias_table,
                                 float* __restrict__ dst) {
  int idx = blockIdx.x * 256 + threadIdx.x;
  if (idx >= 12 * 4 * 4 * 8 * 32) return;
  int lane = idx & 31;
  int jv = (idx >> 5) & 7;
  int nt = (idx >> 8) & 3;
  int mt = (idx >> 10) & 3;
  int h  = idx >> 12;
  int m = mt * 16 + jv + 8 * (lane >> 4);
  int n = nt * 16 + (lane & 15);
  float v = 0.f;
  if (m < NTOK && n < NTOK) {
    int r1 = m / 7, c1 = m % 7, r2 = n / 7, c2 = n % 7;
    int rel = (r1 - r2 + 6) * 13 + (c1 - c2 + 6);
    v = bias_table[rel * HEADS + h];
  }
  dst[idx] = v;
}

// ---------------- fused window attention ----------------------------------
// 1 workgroup = 1 window (49 tokens x 384 ch). 384 threads = 12 waves.
// LN -> qkv GEMM (WMMA) -> per-head attention (WMMA + fragment softmax)
// -> proj GEMM (WMMA) -> +shortcut -> seq[b][hw][c]

#define XSH_P 388   // fp32 pitch
#define XN_P  392   // bf16 pitch (784B, lane stride 196 dwords -> clean banks)
#define QK_P  40    // bf16 pitch for 64x32 q/k tiles
#define VT_P  72    // bf16 pitch for 32x64 v-transposed
#define P_P   72    // bf16 pitch for 64x64 probability tile

__global__ __launch_bounds__(384, 1)
void attn_kernel(const float* __restrict__ x,
                 const float* __restrict__ ln_g, const float* __restrict__ ln_b,
                 const float* __restrict__ qkv_b, const float* __restrict__ proj_b,
                 const bf16* __restrict__ qkvw, const bf16* __restrict__ projw,
                 const float* __restrict__ biasfrag,
                 float* __restrict__ seq) {
  __shared__ float xsh[NTOK * XSH_P];          // 76048 B  shortcut window
  __shared__ bf16  xn[64 * XN_P];              // 50176 B  LN out, reused as attn-out
  __shared__ bf16  qkbuf[HEADS * 2 * 64 * QK_P]; // 122880 B q|k, reused as P
  __shared__ bf16  vtbuf[HEADS * HD * VT_P];   // 55296 B  V transposed [d][n]

  const int tid = threadIdx.x;
  const int lane = tid & 31;
  const int wv = tid >> 5;
  const int wid = blockIdx.x;
  const int b = wid >> 6;
  const int wr = (wid >> 3) & 7;
  const int wc = wid & 7;

  // ---- Phase A: load window + zero pads --------------------------------
  for (int idx = tid; idx < NTOK * DIM; idx += 384) {
    int c = idx / NTOK;
    int n = idx % NTOK;
    int hw = (wr * 7 + n / 7) * 56 + (wc * 7 + n % 7);
    xsh[n * XSH_P + c] = x[((size_t)(b * DIM + c)) * 3136 + hw];
  }
  for (int idx = tid; idx < 15 * XN_P; idx += 384)
    xn[NTOK * XN_P + idx] = (bf16)0.f;   // zero pad rows 49..63 of A matrix
  __syncthreads();

  // ---- LayerNorm (wave per token) --------------------------------------
  for (int n = wv; n < NTOK; n += 12) {
    float vals[12];
    float s = 0.f, s2 = 0.f;
#pragma unroll
    for (int k = 0; k < 12; k++) {
      float v = xsh[n * XSH_P + lane + 32 * k];
      vals[k] = v; s += v; s2 += v * v;
    }
#pragma unroll
    for (int off = 16; off >= 1; off >>= 1) {
      s  += __shfl_xor(s, off, 32);
      s2 += __shfl_xor(s2, off, 32);
    }
    float mu = s * (1.f / 384.f);
    float rstd = rsqrtf(s2 * (1.f / 384.f) - mu * mu + 1e-5f);
#pragma unroll
    for (int k = 0; k < 12; k++) {
      int c = lane + 32 * k;
      xn[n * XN_P + c] = (bf16)((vals[k] - mu) * rstd * ln_g[c] + ln_b[c]);
    }
  }
  __syncthreads();

  const int mrow = lane & 15;
  const int koffA = (lane >> 4) * 8;
  const int koffB = (lane >> 4) * 8;

  // ---- Phase B: qkv GEMM (49x384 @ 384x1152) ---------------------------
  for (int nti = 0; nti < 6; nti++) {
    int nt = wv * 6 + nti;                 // 72 n-tiles over 12 waves
    int j = nt * 16 + mrow;                // output column of this lane
    const bf16* wrowp = qkvw + (size_t)j * DIM;
    v8f acc[4];
#pragma unroll
    for (int mt = 0; mt < 4; mt++) acc[mt] = vzero();
    for (int ks = 0; ks < 12; ks++) {
      v16bf bfg = loadB(wrowp + ks * 32 + koffB);
#pragma unroll
      for (int mt = 0; mt < 4; mt++) {
        v16bf afg = loadA(xn + (mt * 16 + mrow) * XN_P + ks * 32 + koffA);
        acc[mt] = wmma_bf16(afg, bfg, acc[mt]);
      }
    }
    float bias = qkv_b[j];
    int which = nt / 24;                   // 0=q,1=k,2=v (uniform per nt)
    int h = (nt % 24) >> 1;
    int d = (nt & 1) * 16 + mrow;
#pragma unroll
    for (int mt = 0; mt < 4; mt++) {
#pragma unroll
      for (int jv = 0; jv < 8; jv++) {
        int m = mt * 16 + jv + 8 * (lane >> 4);
        float v = acc[mt][jv] + bias;
        if (which == 0)
          qkbuf[h * (2 * 64 * QK_P) + m * QK_P + d] = (bf16)(v * SCALE_ATT);
        else if (which == 1)
          qkbuf[h * (2 * 64 * QK_P) + 64 * QK_P + m * QK_P + d] = (bf16)v;
        else
          vtbuf[h * (HD * VT_P) + d * VT_P + m] = (bf16)v;
      }
    }
  }
  __syncthreads();

  // ---- Phase C: per-head attention (wave = head) -----------------------
  {
    const int h = wv;
    bf16* qb = qkbuf + h * (2 * 64 * QK_P);
    bf16* kb = qb + 64 * QK_P;
    const bf16* vb = vtbuf + h * (HD * VT_P);
    bf16* Pb = qb;  // reuse q|k region for P (64x72 <= 2*64*40)

    v16bf aq[4], bk[4];
#pragma unroll
    for (int t = 0; t < 4; t++) {
      aq[t] = loadA(qb + (t * 16 + mrow) * QK_P + koffA);
      bk[t] = loadB(kb + (t * 16 + mrow) * QK_P + koffB);
    }
#pragma unroll
    for (int mt = 0; mt < 4; mt++) {
      v8f s[4];
#pragma unroll
      for (int nt = 0; nt < 4; nt++) s[nt] = wmma_bf16(aq[mt], bk[nt], vzero());
#pragma unroll
      for (int jv = 0; jv < 8; jv++) {
        float sv[4];
        float mx = -1e30f;
#pragma unroll
        for (int nt = 0; nt < 4; nt++) {
          int n = nt * 16 + mrow;
          float v = s[nt][jv] +
                    biasfrag[(((h * 4 + mt) * 4 + nt) * 8 + jv) * 32 + lane];
          if (n >= NTOK) v = -1e30f;
          sv[nt] = v;
          mx = fmaxf(mx, v);
        }
#pragma unroll
        for (int off = 1; off <= 8; off <<= 1)
          mx = fmaxf(mx, __shfl_xor(mx, off, 32));   // within 16-lane half
        float sum = 0.f;
#pragma unroll
        for (int nt = 0; nt < 4; nt++) { sv[nt] = __expf(sv[nt] - mx); sum += sv[nt]; }
#pragma unroll
        for (int off = 1; off <= 8; off <<= 1)
          sum += __shfl_xor(sum, off, 32);
        float inv = 1.f / sum;
        int m = mt * 16 + jv + 8 * (lane >> 4);
#pragma unroll
        for (int nt = 0; nt < 4; nt++) {
          int n = nt * 16 + mrow;
          Pb[m * P_P + n] = (bf16)((n < NTOK) ? sv[nt] * inv : 0.f);
        }
      }
    }
    // out = P @ V  (same wave, LDS ops in order -> no barrier needed)
#pragma unroll
    for (int mt = 0; mt < 4; mt++) {
      v8f o[2];
      o[0] = vzero(); o[1] = vzero();
#pragma unroll
      for (int kk = 0; kk < 2; kk++) {
        v16bf afg = loadA(Pb + (mt * 16 + mrow) * P_P + kk * 32 + koffA);
#pragma unroll
        for (int dt = 0; dt < 2; dt++) {
          v16bf bfg = loadB(vb + (dt * 16 + mrow) * VT_P + kk * 32 + koffB);
          o[dt] = wmma_bf16(afg, bfg, o[dt]);
        }
      }
#pragma unroll
      for (int dt = 0; dt < 2; dt++)
#pragma unroll
        for (int jv = 0; jv < 8; jv++) {
          int m = mt * 16 + jv + 8 * (lane >> 4);
          xn[m * XN_P + h * HD + dt * 16 + mrow] = (bf16)o[dt][jv];
        }
    }
  }
  __syncthreads();

  // ---- Phase D: proj GEMM + shortcut -> seq ----------------------------
  for (int nti = 0; nti < 2; nti++) {
    int nt = wv * 2 + nti;                 // 24 n-tiles over 12 waves
    int j = nt * 16 + mrow;
    const bf16* wrowp = projw + (size_t)j * DIM;
    v8f acc[4];
#pragma unroll
    for (int mt = 0; mt < 4; mt++) acc[mt] = vzero();
    for (int ks = 0; ks < 12; ks++) {
      v16bf bfg = loadB(wrowp + ks * 32 + koffB);
#pragma unroll
      for (int mt = 0; mt < 4; mt++) {
        v16bf afg = loadA(xn + (mt * 16 + mrow) * XN_P + ks * 32 + koffA);
        acc[mt] = wmma_bf16(afg, bfg, acc[mt]);
      }
    }
    float pb = proj_b[j];
#pragma unroll
    for (int mt = 0; mt < 4; mt++)
#pragma unroll
      for (int jv = 0; jv < 8; jv++) {
        int m = mt * 16 + jv + 8 * (lane >> 4);
        if (m < NTOK) {
          int hw = (wr * 7 + m / 7) * 56 + (wc * 7 + m % 7);
          seq[((size_t)b * 3136 + hw) * DIM + j] =
              acc[mt][jv] + pb + xsh[m * XSH_P + j];
        }
      }
  }
}

// ---------------- depthwise 3x3 conv + BN ---------------------------------

__global__ __launch_bounds__(384, 2)
void conv_bn_kernel(const float* __restrict__ seq, const float* __restrict__ conv_w,
                    const float* __restrict__ bn_g, const float* __restrict__ bn_b,
                    const float* __restrict__ bn_mean, const float* __restrict__ bn_var,
                    float* __restrict__ seq2) {
  int bh = blockIdx.x;
  int b = bh / 56, h = bh % 56;
  int c = threadIdx.x;
  float cw[9];
#pragma unroll
  for (int t = 0; t < 9; t++) cw[t] = conv_w[c * 9 + t];
  float sc = bn_g[c] * rsqrtf(bn_var[c] + 1e-5f);
  float sh = bn_b[c] - bn_mean[c] * sc;
  const float* base = seq + (size_t)b * 3136 * DIM;
  float* obase = seq2 + (size_t)b * 3136 * DIM;
  for (int w = 0; w < 56; w++) {
    float acc = 0.f;
#pragma unroll
    for (int dh = -1; dh <= 1; dh++) {
      int hh = h + dh;
      if ((unsigned)hh < 56u) {
#pragma unroll
        for (int dw = -1; dw <= 1; dw++) {
          int ww = w + dw;
          if ((unsigned)ww < 56u)
            acc += base[(size_t)(hh * 56 + ww) * DIM + c] * cw[(dh + 1) * 3 + dw + 1];
        }
      }
    }
    obase[(size_t)(h * 56 + w) * DIM + c] = acc * sc + sh;
  }
}

// ---------------- fused MLP (GEMM1 -> GELU -> GEMM2 -> residual) ----------
// 64 rows / workgroup, 256 threads = 8 waves. Output transposed via LDS for
// coalesced NCHW stores.

#define IN_P 392
#define H_P  1544
#define OUT_P 388

__global__ __launch_bounds__(256, 1)
void mlp_kernel(const float* __restrict__ seq2,
                const bf16* __restrict__ w1b, const float* __restrict__ b1,
                const bf16* __restrict__ w2b, const float* __restrict__ b2,
                float* __restrict__ out) {
  __shared__ bf16 inb[64 * IN_P];                 // 50176 B
  __shared__ __align__(16) char hmem[64 * H_P * 2]; // 197632 B (hbuf / outb alias)
  bf16* hbuf = (bf16*)hmem;
  float* outb = (float*)hmem;

  const int tid = threadIdx.x;
  const int lane = tid & 31;
  const int wv = tid >> 5;
  const size_t rowBase = (size_t)blockIdx.x * 64;
  const int b = (int)(rowBase / 3136);
  const int hw0 = (int)(rowBase % 3136);          // 3136 % 64 == 0: no crossing

  for (int idx = tid; idx < 64 * DIM; idx += 256) {
    int m = idx / DIM, c = idx % DIM;
    inb[m * IN_P + c] = (bf16)seq2[(rowBase + m) * DIM + c];
  }
  __syncthreads();

  const int mrow = lane & 15;
  const int koffA = (lane >> 4) * 8;
  const int koffB = (lane >> 4) * 8;

  // GEMM1 (64x384 @ 384x1536) + GELU
  for (int nti = 0; nti < 12; nti++) {
    int nt = wv * 12 + nti;
    int j = nt * 16 + mrow;
    const bf16* wrowp = w1b + (size_t)j * DIM;
    v8f acc[4];
#pragma unroll
    for (int mt = 0; mt < 4; mt++) acc[mt] = vzero();
    for (int ks = 0; ks < 12; ks++) {
      v16bf bfg = loadB(wrowp + ks * 32 + koffB);
#pragma unroll
      for (int mt = 0; mt < 4; mt++) {
        v16bf afg = loadA(inb + (mt * 16 + mrow) * IN_P + ks * 32 + koffA);
        acc[mt] = wmma_bf16(afg, bfg, acc[mt]);
      }
    }
    float bias = b1[j];
#pragma unroll
    for (int mt = 0; mt < 4; mt++)
#pragma unroll
      for (int jv = 0; jv < 8; jv++) {
        int m = mt * 16 + jv + 8 * (lane >> 4);
        float v = acc[mt][jv] + bias;
        v = 0.5f * v * (1.f + erff(v * 0.70710678f));   // exact GELU
        hbuf[m * H_P + j] = (bf16)v;
      }
  }
  __syncthreads();

  // GEMM2 (64x1536 @ 1536x384), keep all accs in regs before aliasing hbuf
  v8f acc2[3][4];
#pragma unroll
  for (int i = 0; i < 3; i++)
#pragma unroll
    for (int mt = 0; mt < 4; mt++) acc2[i][mt] = vzero();
  for (int nti = 0; nti < 3; nti++) {
    int nt = wv * 3 + nti;
    int j = nt * 16 + mrow;
    const bf16* wrowp = w2b + (size_t)j * HIDDEN;
    for (int ks = 0; ks < 48; ks++) {
      v16bf bfg = loadB(wrowp + ks * 32 + koffB);
#pragma unroll
      for (int mt = 0; mt < 4; mt++) {
        v16bf afg = loadA(hbuf + (mt * 16 + mrow) * H_P + ks * 32 + koffA);
        acc2[nti][mt] = wmma_bf16(afg, bfg, acc2[nti][mt]);
      }
    }
  }
  __syncthreads();   // all hbuf reads done -> safe to alias as outb

#pragma unroll
  for (int nti = 0; nti < 3; nti++) {
    int j = (wv * 3 + nti) * 16 + mrow;
    float bias = b2[j];
#pragma unroll
    for (int mt = 0; mt < 4; mt++)
#pragma unroll
      for (int jv = 0; jv < 8; jv++) {
        int m = mt * 16 + jv + 8 * (lane >> 4);
        outb[m * OUT_P + j] = acc2[nti][mt][jv] + bias +
                              seq2[(rowBase + m) * DIM + j];
      }
  }
  __syncthreads();

  // coalesced NCHW writeout: 64 consecutive hw per channel
  for (int idx = tid; idx < 64 * DIM; idx += 256) {
    int c = idx / 64, i = idx % 64;
    out[((size_t)b * DIM + c) * 3136 + hw0 + i] = outb[i * OUT_P + c];
  }
}

// ---------------- launch ---------------------------------------------------

extern "C" void kernel_launch(void* const* d_in, const int* in_sizes, int n_in,
                              void* d_out, int out_size, void* d_ws, size_t ws_size,
                              hipStream_t stream) {
  const float* x        = (const float*)d_in[0];
  const float* ln_g     = (const float*)d_in[1];
  const float* ln_b     = (const float*)d_in[2];
  const float* qkv_w    = (const float*)d_in[3];
  const float* qkv_b    = (const float*)d_in[4];
  const float* proj_w   = (const float*)d_in[5];
  const float* proj_b   = (const float*)d_in[6];
  const float* bias_tab = (const float*)d_in[7];
  const float* conv_w   = (const float*)d_in[8];
  const float* bn_g     = (const float*)d_in[9];
  const float* bn_b     = (const float*)d_in[10];
  const float* bn_mean  = (const float*)d_in[11];
  const float* bn_var   = (const float*)d_in[12];
  const float* w1       = (const float*)d_in[13];
  const float* b1       = (const float*)d_in[14];
  const float* w2       = (const float*)d_in[15];
  const float* b2       = (const float*)d_in[16];
  float* outp = (float*)d_out;

  char* ws = (char*)d_ws;
  const size_t SEQ_BYTES = (size_t)32 * 3136 * DIM * 4;   // 154140672
  float* seq     = (float*)(ws);
  float* seq2    = (float*)(ws + SEQ_BYTES);
  bf16*  qkvw_b  = (bf16*)(ws + 2 * SEQ_BYTES);
  bf16*  projw_b = (bf16*)((char*)qkvw_b  + (size_t)1152 * 384 * 2);
  bf16*  w1_b    = (bf16*)((char*)projw_b + (size_t)384 * 384 * 2);
  bf16*  w2_b    = (bf16*)((char*)w1_b    + (size_t)1536 * 384 * 2);
  float* biasfr  = (float*)((char*)w2_b   + (size_t)384 * 1536 * 2);

  int n;
  n = 1152 * 384; cvt_bf16_kernel<<<(n + 255) / 256, 256, 0, stream>>>(qkv_w, qkvw_b, n);
  n = 384 * 384;  cvt_bf16_kernel<<<(n + 255) / 256, 256, 0, stream>>>(proj_w, projw_b, n);
  n = 1536 * 384; cvt_bf16_kernel<<<(n + 255) / 256, 256, 0, stream>>>(w1, w1_b, n);
  n = 384 * 1536; cvt_bf16_kernel<<<(n + 255) / 256, 256, 0, stream>>>(w2, w2_b, n);
  n = 12 * 4 * 4 * 8 * 32;
  bias_frag_kernel<<<(n + 255) / 256, 256, 0, stream>>>(bias_tab, biasfr);

  attn_kernel<<<2048, 384, 0, stream>>>(x, ln_g, ln_b, qkv_b, proj_b,
                                        qkvw_b, projw_b, biasfr, seq);
  conv_bn_kernel<<<32 * 56, 384, 0, stream>>>(seq, conv_w, bn_g, bn_b,
                                              bn_mean, bn_var, seq2);
  mlp_kernel<<<32 * 3136 / 64, 256, 0, stream>>>(seq2, w1_b, b1, w2_b, b2, outp);
}
